// DecoderBlock_19825569038501
// MI455X (gfx1250) — compile-verified, matrix operand order checked by reading, not run on hardware
//
#include <hip/hip_runtime.h>
#include <math.h>

// ---------------------------------------------------------------------------
// Types for CDNA5 WMMA (gfx1250, wave32)
// ---------------------------------------------------------------------------
typedef __attribute__((ext_vector_type(16))) __bf16 v16bf;
typedef __attribute__((ext_vector_type(8)))  __bf16 v8bf;
typedef __attribute__((ext_vector_type(8)))  float  v8f;
typedef int vi4 __attribute__((vector_size(16)));   // 16-byte granule for async DMA

static_assert(sizeof(v16bf) == 32, "v16bf must be 8 VGPRs");
static_assert(sizeof(v8f)   == 32, "v8f must be 8 VGPRs");

__device__ __forceinline__ v16bf mk16(v8bf lo, v8bf hi) {
  return __builtin_shufflevector(lo, hi, 0,1,2,3,4,5,6,7,8,9,10,11,12,13,14,15);
}

__device__ __forceinline__ v8f wmma_bf16(v16bf a, v16bf b, v8f c) {
  // D = A(16x32 bf16) * B(32x16 bf16) + C(16x16 f32)
  return __builtin_amdgcn_wmma_f32_16x16x32_bf16(false, a, false, b, (short)0, c,
                                                 false, false);
}

__device__ __forceinline__ v8f zero8() {
  v8f z = {0.f, 0.f, 0.f, 0.f, 0.f, 0.f, 0.f, 0.f};
  return z;
}

// ---------------------------------------------------------------------------
// CDNA5 async global->LDS copy (GLOBAL_LOAD_ASYNC_TO_LDS_B128, ASYNCcnt).
// Builtin prototype (from hipcc diagnostic): (vi4 addrspace(1)*, vi4 addrspace(3)*,
// imm offset, cpol).  Falls back to a VGPR round-trip copy if absent.
// ---------------------------------------------------------------------------
#if __has_builtin(__builtin_amdgcn_global_load_async_to_lds_b128)
#define HAVE_ASYNC_COPY 1
#endif

__device__ __forceinline__ void async_b128(const __bf16* g, __bf16* l) {
#if defined(HAVE_ASYNC_COPY)
  __builtin_amdgcn_global_load_async_to_lds_b128(
      (__attribute__((address_space(1))) vi4*)g,
      (__attribute__((address_space(3))) vi4*)l, 0, 0);
#else
  *(v8bf*)l = *(const v8bf*)g;
#endif
}

__device__ __forceinline__ void wait_async() {
#if defined(HAVE_ASYNC_COPY)
#if __has_builtin(__builtin_amdgcn_s_wait_asynccnt)
  __builtin_amdgcn_s_wait_asynccnt(0);
#else
  asm volatile("s_wait_asynccnt 0x0" ::: "memory");
#endif
#endif
}

// ---------------------------------------------------------------------------
// Model constants
// ---------------------------------------------------------------------------
constexpr int NB    = 2;      // batch
constexpr int SEQ   = 2048;   // sequence length (q and k for both attentions)
constexpr int DEMB  = 1024;
constexpr int HEADS = 16;
constexpr int HDIM  = 64;     // DEMB / HEADS
constexpr int DFF   = 4096;
constexpr int TOK   = NB * SEQ;   // 4096 total tokens

// ---------------------------------------------------------------------------
// fp32 -> bf16 conversion
// ---------------------------------------------------------------------------
__global__ __launch_bounds__(256) void cvt_f32_bf16(const float* __restrict__ x,
                                                    __bf16* __restrict__ y, int n) {
  int i = blockIdx.x * 256 + threadIdx.x;
  if (i < n) y[i] = (__bf16)x[i];
}

// ---------------------------------------------------------------------------
// Tiled bf16 GEMM with WMMA:  C[M,N] = A[M,K] @ B[K,N] + bias
//  block tile 128x128, 8 waves (2x4), wave tile 64x32 (4x2 WMMA frags), BK=32
//  Double-buffered LDS: A staged by async DMA, B register-prefetched+transposed.
//  MODE: 0 = bf16 out, 1 = f32 out, 2 = bf16 out + ReLU
// ---------------------------------------------------------------------------
template <int MODE>
__global__ __launch_bounds__(256)
void gemm_wmma_bf16(const __bf16* __restrict__ A, const __bf16* __restrict__ B,
                    const float* __restrict__ bias, void* __restrict__ Cout,
                    int M, int N, int K) {
  __shared__ __bf16 Alds[2][128][40];   // [m][k], padded
  __shared__ __bf16 Blds[2][128][40];   // [n][k] (transposed), padded

  const int tid    = threadIdx.x;
  const int lane   = tid & 31;
  const int wave   = tid >> 5;       // 0..7
  const int wm     = wave >> 2;      // 0..1  (M direction, 64 rows each)
  const int wn     = wave & 3;       // 0..3  (N direction, 32 cols each)
  const int l15    = lane & 15;
  const int half   = lane >> 4;      // 0|1
  const int kb     = half * 8;       // fragment K base (A/B 16-bit layout)
  const int rowoff = half * 8;       // C/D fragment row base

  const int m0 = blockIdx.y * 128;
  const int n0 = blockIdx.x * 128;

  // staging coordinates
  const int ar = tid >> 2;           // 0..63   (A: two rows per thread)
  const int ac = (tid & 3) * 8;      // 0,8,16,24
  const int bk = tid >> 4;           // 0..15   (B: two k-rows per thread)
  const int bn = (tid & 15) * 8;     // 0..120

  auto stageA = [&](int k0, int buf) {
#pragma unroll
    for (int rr = ar; rr < 128; rr += 64)
      async_b128(A + (size_t)(m0 + rr) * K + k0 + ac, &Alds[buf][rr][ac]);
  };
  auto loadB = [&](int k0, v8bf breg[2]) {
#pragma unroll
    for (int i = 0; i < 2; ++i)
      breg[i] = *(const v8bf*)(B + (size_t)(k0 + bk + 16 * i) * N + n0 + bn);
  };
  auto storeB = [&](const v8bf breg[2], int buf) {
#pragma unroll
    for (int i = 0; i < 2; ++i)
#pragma unroll
      for (int j = 0; j < 8; ++j) Blds[buf][bn + j][bk + 16 * i] = breg[i][j];
  };

  v8f acc[4][2];
#pragma unroll
  for (int mi = 0; mi < 4; ++mi)
#pragma unroll
    for (int ni = 0; ni < 2; ++ni) acc[mi][ni] = zero8();

  const int nk = K / 32;

  // ---- prologue: stage tile 0 ----
  {
    v8bf breg[2];
    stageA(0, 0);
    loadB(0, breg);
    storeB(breg, 0);
  }

  for (int t = 0; t < nk; ++t) {
    const int buf = t & 1;
    wait_async();          // tile t's async A copies done (tile t+1 not yet issued)
    __syncthreads();       // all waves' tile-t staging visible

    v8bf bnext[2];
    const bool hn = (t + 1) < nk;
    if (hn) {
      stageA((t + 1) * 32, buf ^ 1);     // async DMA overlaps compute below
      loadB((t + 1) * 32, bnext);        // global latency overlaps compute
      if (t + 2 < nk)
        __builtin_prefetch(B + (size_t)((t + 2) * 32 + bk) * N + n0 + bn, 0, 0);
    }

    // ---- compute tile t ----
    v16bf bfr[2];
#pragma unroll
    for (int ni = 0; ni < 2; ++ni) {
      const __bf16* br = &Blds[buf][wn * 32 + ni * 16 + l15][0];
      bfr[ni] = mk16(*(const v8bf*)(br + kb), *(const v8bf*)(br + kb + 16));
    }
#pragma unroll
    for (int mi = 0; mi < 4; ++mi) {
      const __bf16* arp = &Alds[buf][wm * 64 + mi * 16 + l15][0];
      v16bf af = mk16(*(const v8bf*)(arp + kb), *(const v8bf*)(arp + kb + 16));
#pragma unroll
      for (int ni = 0; ni < 2; ++ni) acc[mi][ni] = wmma_bf16(af, bfr[ni], acc[mi][ni]);
    }

    if (hn) storeB(bnext, buf ^ 1);      // loadcnt wait deferred to here
  }

  // ---- epilogue: bias (+ReLU) and store ----
#pragma unroll
  for (int mi = 0; mi < 4; ++mi) {
#pragma unroll
    for (int ni = 0; ni < 2; ++ni) {
      int col = n0 + wn * 32 + ni * 16 + l15;
      float bv = bias[col];
#pragma unroll
      for (int r = 0; r < 8; ++r) {
        int row = m0 + wm * 64 + mi * 16 + rowoff + r;
        float v = acc[mi][ni][r] + bv;
        if (MODE == 2) v = fmaxf(v, 0.f);
        if (MODE == 1)
          ((float*)Cout)[(size_t)row * N + col] = v;
        else
          ((__bf16*)Cout)[(size_t)row * N + col] = (__bf16)v;
      }
    }
  }
}

// ---------------------------------------------------------------------------
// Flash-style attention for one decoder block MHA.
//  grid: (SEQ/128, HEADS, NB), block: 256 threads = 8 waves.
//  Each wave owns 16 q-rows; streams 64-key tiles of K/V through a
//  double-buffered LDS pipeline (K via async DMA, V register-prefetched).
//  MASKMODE 0: mask[N,1,Q,K] (self-attn trg_mask); 1: mask[N,1,1,K] (src_mask)
// ---------------------------------------------------------------------------
template <int MASKMODE>
__global__ __launch_bounds__(256)
void attn_wmma(const __bf16* __restrict__ Qb, const __bf16* __restrict__ Kb,
               const __bf16* __restrict__ Vb, const int* __restrict__ mask,
               __bf16* __restrict__ Ob) {
  __shared__ __bf16 Klds[2][64][72];    // [key][d]  (row-major == B-frag layout)
  __shared__ __bf16 Vlds[2][64][72];    // [d][key]  (transposed)
  __shared__ __bf16 Plds[8][16][72];    // per-wave P tile for C->A layout fixup

  const int tid    = threadIdx.x;
  const int lane   = tid & 31;
  const int wave   = tid >> 5;
  const int l15    = lane & 15;
  const int half   = lane >> 4;
  const int kb     = half * 8;
  const int rowoff = half * 8;

  const int n  = blockIdx.z;
  const int h  = blockIdx.y;
  const int q0 = blockIdx.x * 128;
  const int qw = q0 + wave * 16;              // wave's first q row
  const float scale = 0.125f;                 // 1/sqrt(64)

  // staging coordinates (K/V tiles are 64 keys x 64 d)
  const int sr = tid >> 3;             // 0..31  (two key rows per thread)
  const int sc = (tid & 7) * 8;        // 0..56

  auto stageK = [&](int k0, int buf) {
#pragma unroll
    for (int rr = sr; rr < 64; rr += 32)
      async_b128(Kb + (size_t)(n * SEQ + k0 + rr) * DEMB + h * HDIM + sc,
                 &Klds[buf][rr][sc]);
  };
  auto loadV = [&](int k0, v8bf vreg[2]) {
#pragma unroll
    for (int i = 0; i < 2; ++i)
      vreg[i] = *(const v8bf*)(Vb + (size_t)(n * SEQ + k0 + sr + 32 * i) * DEMB +
                               h * HDIM + sc);
  };
  auto storeV = [&](const v8bf vreg[2], int buf) {
#pragma unroll
    for (int i = 0; i < 2; ++i)
#pragma unroll
      for (int j = 0; j < 8; ++j) Vlds[buf][sc + j][sr + 32 * i] = vreg[i][j];
  };

  // ---- Q fragments (2 x 16x32 over head dim), loaded straight from global ----
  const __bf16* qrow = Qb + (size_t)(n * SEQ + qw + l15) * DEMB + h * HDIM;
  v16bf qf0 = mk16(*(const v8bf*)(qrow + kb),      *(const v8bf*)(qrow + kb + 16));
  v16bf qf1 = mk16(*(const v8bf*)(qrow + 32 + kb), *(const v8bf*)(qrow + 48 + kb));

  v8f Oacc[4];
#pragma unroll
  for (int f = 0; f < 4; ++f) Oacc[f] = zero8();
  float m_r[8], l_r[8];
#pragma unroll
  for (int r = 0; r < 8; ++r) { m_r[r] = -INFINITY; l_r[r] = 0.f; }

  const int nk = SEQ / 64;

  // ---- prologue: stage tile 0 ----
  {
    v8bf vreg[2];
    stageK(0, 0);
    loadV(0, vreg);
    storeV(vreg, 0);
  }

  for (int t = 0; t < nk; ++t) {
    const int buf = t & 1;
    const int k0 = t * 64;
    wait_async();
    __syncthreads();

    v8bf vnext[2];
    const bool hn = (t + 1) < nk;
    if (hn) {
      stageK((t + 1) * 64, buf ^ 1);
      loadV((t + 1) * 64, vnext);
    }

    // ---- S = scale * Q K^T  (4 key sub-tiles of 16), mask, online softmax ----
    float sv[4][8];
#pragma unroll
    for (int f = 0; f < 4; ++f) {
      v8f acc = zero8();
      const __bf16* kr = &Klds[buf][f * 16 + l15][0];
      v16bf b0 = mk16(*(const v8bf*)(kr + kb),      *(const v8bf*)(kr + kb + 16));
      v16bf b1 = mk16(*(const v8bf*)(kr + 32 + kb), *(const v8bf*)(kr + 48 + kb));
      acc = wmma_bf16(qf0, b0, acc);
      acc = wmma_bf16(qf1, b1, acc);
      const int kcol = k0 + f * 16 + l15;
      if (MASKMODE == 1) {
        const int mk = mask[(size_t)n * SEQ + kcol];
#pragma unroll
        for (int r = 0; r < 8; ++r) {
          float v = acc[r] * scale;
          sv[f][r] = (mk == 0) ? -INFINITY : v;
        }
      } else {
#pragma unroll
        for (int r = 0; r < 8; ++r) {
          const int q = qw + rowoff + r;
          const int mk = mask[((size_t)n * SEQ + q) * SEQ + kcol];
          float v = acc[r] * scale;
          sv[f][r] = (mk == 0) ? -INFINITY : v;
        }
      }
    }

    float mnew[8], rsum[8];
#pragma unroll
    for (int r = 0; r < 8; ++r) {
      float mx = m_r[r];
#pragma unroll
      for (int f = 0; f < 4; ++f) mx = fmaxf(mx, sv[f][r]);
#pragma unroll
      for (int off = 8; off > 0; off >>= 1) mx = fmaxf(mx, __shfl_xor(mx, off));
      mnew[r] = mx;
    }
#pragma unroll
    for (int r = 0; r < 8; ++r) {
      float s = 0.f;
#pragma unroll
      for (int f = 0; f < 4; ++f) {
        float p = __expf(sv[f][r] - mnew[r]);
        sv[f][r] = p;
        s += p;
      }
#pragma unroll
      for (int off = 8; off > 0; off >>= 1) s += __shfl_xor(s, off);
      rsum[r] = s;
    }
#pragma unroll
    for (int r = 0; r < 8; ++r) {
      float corr = __expf(m_r[r] - mnew[r]);   // exp(-inf - finite) = 0 first iter
      l_r[r] = l_r[r] * corr + rsum[r];
      m_r[r] = mnew[r];
#pragma unroll
      for (int f = 0; f < 4; ++f) Oacc[f][r] *= corr;
    }

    // ---- C-layout P -> per-wave LDS -> A-layout fragments ----
#pragma unroll
    for (int f = 0; f < 4; ++f)
#pragma unroll
      for (int r = 0; r < 8; ++r)
        Plds[wave][rowoff + r][f * 16 + l15] = (__bf16)sv[f][r];

    const __bf16* pr = &Plds[wave][l15][0];
    v16bf p0 = mk16(*(const v8bf*)(pr + kb),      *(const v8bf*)(pr + kb + 16));
    v16bf p1 = mk16(*(const v8bf*)(pr + 32 + kb), *(const v8bf*)(pr + 48 + kb));

    // ---- O += P @ V  (4 d sub-tiles of 16) ----
#pragma unroll
    for (int f = 0; f < 4; ++f) {
      const __bf16* vr = &Vlds[buf][f * 16 + l15][0];
      v16bf vb0 = mk16(*(const v8bf*)(vr + kb),      *(const v8bf*)(vr + kb + 16));
      v16bf vb1 = mk16(*(const v8bf*)(vr + 32 + kb), *(const v8bf*)(vr + 48 + kb));
      Oacc[f] = wmma_bf16(p0, vb0, Oacc[f]);
      Oacc[f] = wmma_bf16(p1, vb1, Oacc[f]);
    }

    if (hn) storeV(vnext, buf ^ 1);      // loadcnt wait deferred to here
  }

  // ---- finalize: O /= l, store bf16 context [token, h*64 + d] ----
#pragma unroll
  for (int f = 0; f < 4; ++f) {
#pragma unroll
    for (int r = 0; r < 8; ++r) {
      const int q = qw + rowoff + r;
      float o = Oacc[f][r] / l_r[r];
      Ob[(size_t)(n * SEQ + q) * DEMB + h * HDIM + f * 16 + l15] = (__bf16)o;
    }
  }
}

// ---------------------------------------------------------------------------
// Residual + LayerNorm over D=1024.  One block (256 thr) per row.
//  Yf = LN(A+B)*g + b (fp32);  optionally also bf16 copy for the next GEMM.
// ---------------------------------------------------------------------------
template <bool WB>
__global__ __launch_bounds__(256)
void ln_residual(const float* __restrict__ A, const float* __restrict__ B,
                 const float* __restrict__ g, const float* __restrict__ bb,
                 float* __restrict__ Yf, __bf16* __restrict__ Ybf) {
  const int row = blockIdx.x;
  const int tid = threadIdx.x;
  const float* a = A + (size_t)row * DEMB;
  const float* b = B + (size_t)row * DEMB;

  float v[4];
  float s = 0.f;
#pragma unroll
  for (int i = 0; i < 4; ++i) {
    int c = tid + i * 256;
    v[i] = a[c] + b[c];
    s += v[i];
  }
  __shared__ float red[8];
#pragma unroll
  for (int off = 16; off > 0; off >>= 1) s += __shfl_xor(s, off);
  if ((tid & 31) == 0) red[tid >> 5] = s;
  __syncthreads();
  float mu = (red[0] + red[1] + red[2] + red[3] + red[4] + red[5] + red[6] + red[7])
             * (1.f / DEMB);

  float vs = 0.f;
#pragma unroll
  for (int i = 0; i < 4; ++i) {
    float d = v[i] - mu;
    vs += d * d;
  }
#pragma unroll
  for (int off = 16; off > 0; off >>= 1) vs += __shfl_xor(vs, off);
  __syncthreads();
  if ((tid & 31) == 0) red[tid >> 5] = vs;
  __syncthreads();
  float var = (red[0] + red[1] + red[2] + red[3] + red[4] + red[5] + red[6] + red[7])
              * (1.f / DEMB);
  float inv = rsqrtf(var + 1e-5f);

#pragma unroll
  for (int i = 0; i < 4; ++i) {
    int c = tid + i * 256;
    float y = (v[i] - mu) * inv * g[c] + bb[c];
    Yf[(size_t)row * DEMB + c] = y;
    if constexpr (WB) Ybf[(size_t)row * DEMB + c] = (__bf16)y;
  }
}

// ---------------------------------------------------------------------------
// Host-side orchestration
// ---------------------------------------------------------------------------
extern "C" void kernel_launch(void* const* d_in, const int* in_sizes, int n_in,
                              void* d_out, int out_size, void* d_ws, size_t ws_size,
                              hipStream_t stream) {
  (void)in_sizes; (void)n_in; (void)out_size; (void)ws_size;

  const float* trg      = (const float*)d_in[0];
  const float* src      = (const float*)d_in[1];
  const int*   trg_mask = (const int*)d_in[2];
  const int*   src_mask = (const int*)d_in[3];
  const float* Wq1 = (const float*)d_in[4];  const float* bq1 = (const float*)d_in[5];
  const float* Wo1 = (const float*)d_in[6];  const float* bo1 = (const float*)d_in[7];
  const float* Wq2 = (const float*)d_in[8];  const float* bq2 = (const float*)d_in[9];
  const float* Wo2 = (const float*)d_in[10]; const float* bo2 = (const float*)d_in[11];
  const float* Wff1 = (const float*)d_in[12]; const float* bff1 = (const float*)d_in[13];
  const float* Wff2 = (const float*)d_in[14]; const float* bff2 = (const float*)d_in[15];
  const float* g1 = (const float*)d_in[16]; const float* b1 = (const float*)d_in[17];
  const float* g2 = (const float*)d_in[18]; const float* b2 = (const float*)d_in[19];
  const float* g3 = (const float*)d_in[20]; const float* b3 = (const float*)d_in[21];
  float* out = (float*)d_out;

  // ---- scratch layout ----
  char* ws = (char*)d_ws;
  auto carve = [&](size_t bytes) {
    char* p = ws;
    ws += (bytes + 255) & ~size_t(255);
    return p;
  };
  __bf16* wq1b  = (__bf16*)carve((size_t)DEMB * DEMB * 2);
  __bf16* wo1b  = (__bf16*)carve((size_t)DEMB * DEMB * 2);
  __bf16* wq2b  = (__bf16*)carve((size_t)DEMB * DEMB * 2);
  __bf16* wo2b  = (__bf16*)carve((size_t)DEMB * DEMB * 2);
  __bf16* wff1b = (__bf16*)carve((size_t)DEMB * DFF * 2);
  __bf16* wff2b = (__bf16*)carve((size_t)DFF * DEMB * 2);
  __bf16* a_bf  = (__bf16*)carve((size_t)TOK * DEMB * 2);   // activation (bf16)
  __bf16* s_bf  = (__bf16*)carve((size_t)TOK * DEMB * 2);   // encoded_src (bf16)
  __bf16* p_q   = (__bf16*)carve((size_t)TOK * DEMB * 2);   // q/k/v or q2 projection
  __bf16* p_kv  = (__bf16*)carve((size_t)TOK * DEMB * 2);   // kv2 projection
  __bf16* ctx   = (__bf16*)carve((size_t)TOK * DEMB * 2);   // attention context
  __bf16* h_ff  = (__bf16*)carve((size_t)TOK * DFF * 2);    // FFN hidden
  float*  tmp_f = (float*)carve((size_t)TOK * DEMB * 4);    // GEMM fp32 output
  float*  xA    = (float*)carve((size_t)TOK * DEMB * 4);    // residual stream 1
  float*  xB    = (float*)carve((size_t)TOK * DEMB * 4);    // residual stream 2

  auto cvt = [&](const float* x, __bf16* y, size_t n) {
    cvt_f32_bf16<<<(unsigned)((n + 255) / 256), 256, 0, stream>>>(x, y, (int)n);
  };

  // ---- bf16 conversions ----
  cvt(trg, a_bf, (size_t)TOK * DEMB);
  cvt(src, s_bf, (size_t)TOK * DEMB);
  cvt(Wq1, wq1b, (size_t)DEMB * DEMB);
  cvt(Wo1, wo1b, (size_t)DEMB * DEMB);
  cvt(Wq2, wq2b, (size_t)DEMB * DEMB);
  cvt(Wo2, wo2b, (size_t)DEMB * DEMB);
  cvt(Wff1, wff1b, (size_t)DEMB * DFF);
  cvt(Wff2, wff2b, (size_t)DFF * DEMB);

  const dim3 gD(DEMB / 128, TOK / 128);   // GEMM -> [TOK, DEMB]
  const dim3 gF(DFF / 128, TOK / 128);    // GEMM -> [TOK, DFF]
  const dim3 gA(SEQ / 128, HEADS, NB);    // attention

  // ---- 1) self-attention: shared q/k/v projection (source "bug") ----
  gemm_wmma_bf16<0><<<gD, 256, 0, stream>>>(a_bf, wq1b, bq1, p_q, TOK, DEMB, DEMB);
  attn_wmma<0><<<gA, 256, 0, stream>>>(p_q, p_q, p_q, trg_mask, ctx);
  gemm_wmma_bf16<1><<<gD, 256, 0, stream>>>(ctx, wo1b, bo1, tmp_f, TOK, DEMB, DEMB);
  ln_residual<true><<<TOK, 256, 0, stream>>>(trg, tmp_f, g1, b1, xA, a_bf);

  // ---- 2) cross-attention: q from x1, shared k/v projection of encoded_src ----
  gemm_wmma_bf16<0><<<gD, 256, 0, stream>>>(a_bf, wq2b, bq2, p_q, TOK, DEMB, DEMB);
  gemm_wmma_bf16<0><<<gD, 256, 0, stream>>>(s_bf, wq2b, bq2, p_kv, TOK, DEMB, DEMB);
  attn_wmma<1><<<gA, 256, 0, stream>>>(p_q, p_kv, p_kv, src_mask, ctx);
  gemm_wmma_bf16<1><<<gD, 256, 0, stream>>>(ctx, wo2b, bo2, tmp_f, TOK, DEMB, DEMB);
  ln_residual<true><<<TOK, 256, 0, stream>>>(xA, tmp_f, g2, b2, xB, a_bf);

  // ---- 3) FFN ----
  gemm_wmma_bf16<2><<<gF, 256, 0, stream>>>(a_bf, wff1b, bff1, h_ff, TOK, DFF, DEMB);
  gemm_wmma_bf16<1><<<gD, 256, 0, stream>>>(h_ff, wff2b, bff2, tmp_f, TOK, DEMB, DFF);
  ln_residual<false><<<TOK, 256, 0, stream>>>(xB, tmp_f, g3, b3, out, nullptr);
}
